// GAT_72224170049540
// MI455X (gfx1250) — compile-verified, matrix operand order checked by reading, not run on hardware
//
#include <hip/hip_runtime.h>
#include <cstdint>
#include <utility>

// ---------------- problem constants ----------------
constexpr int N_NODES = 40000;
constexpr int N_EDGES = 60000;
constexpr int HC      = 256;          // H*C = 4*64
constexpr int NC      = 51;
constexpr int NC_PAD  = 64;
constexpr int EN      = N_EDGES + N_NODES;   // edges + self loops
constexpr int TWO_E   = 2 * N_EDGES;
constexpr int SORT_N  = 131072;              // next pow2 >= 2E

// ---------------- CDNA5 WMMA types ----------------
typedef __bf16 bf16_t;
typedef __attribute__((ext_vector_type(8)))  __bf16 v8bf;
typedef __attribute__((ext_vector_type(16))) __bf16 v16bf;
typedef __attribute__((ext_vector_type(8)))  float  v8f;

__device__ __forceinline__ unsigned short f32_to_bf16(float f) {
  unsigned int u = __float_as_uint(f);
  u += 0x7FFFu + ((u >> 16) & 1u);     // round to nearest even
  return (unsigned short)(u >> 16);
}

// Load one 16x32 bf16 fragment half-row pair per lane.
// base points at (row_base + k0) in ushort units; kh = lane>>4.
// elements 0..7  -> K = k0 + kh*8 + 0..7
// elements 8..15 -> K = k0 + 16 + kh*8 + 0..7   (ISA 16-bit A/B layout)
__device__ __forceinline__ v16bf load_frag(const unsigned short* base, int kh) {
  v8bf lo = *(const v8bf*)(base + kh * 8);
  v8bf hi = *(const v8bf*)(base + 16 + kh * 8);
  v16bf f;
#pragma unroll
  for (int i = 0; i < 8; ++i) { f[i] = lo[i]; f[i + 8] = hi[i]; }
  return f;
}

__device__ __forceinline__ void atomicMaxF(float* addr, float val) {
  if (val >= 0.0f) atomicMax((int*)addr, __float_as_int(val));
  else             atomicMin((unsigned int*)addr, __float_as_uint(val));
}

// ---------------- utility kernels ----------------
__global__ void fill_f32_kernel(float* p, float v, long n) {
  long i = (long)blockIdx.x * blockDim.x + threadIdx.x;
  if (i < n) p[i] = v;
}

__global__ void f32_to_bf16_kernel(const float* __restrict__ in,
                                   unsigned short* __restrict__ out, long n) {
  long i = (long)blockIdx.x * blockDim.x + threadIdx.x;
  if (i < n) out[i] = f32_to_bf16(in[i]);
}

// W [256 x 256] row-major f32 -> WT [256 x 256] bf16, WT[n*256+k] = W[k*256+n]
__global__ void transpose_w_kernel(const float* __restrict__ W,
                                   unsigned short* __restrict__ WT) {
  int idx = blockIdx.x * blockDim.x + threadIdx.x;
  if (idx >= 256 * 256) return;
  int n = idx >> 8, k = idx & 255;
  WT[idx] = f32_to_bf16(W[k * 256 + n]);
}

// Wc [1024 x 51] f32 -> WcT [64 x 1024] bf16 (cols 51..63 zero-padded)
__global__ void prep_wct_kernel(const float* __restrict__ Wc,
                                unsigned short* __restrict__ WcT) {
  int idx = blockIdx.x * blockDim.x + threadIdx.x;
  if (idx >= 64 * 1024) return;
  int n = idx >> 10, k = idx & 1023;
  float v = (n < NC) ? Wc[k * NC + n] : 0.0f;
  WcT[idx] = f32_to_bf16(v);
}

// ---------------- WMMA GEMMs ----------------
// hfeat[M x 256] = A(bf16)[M x 256] @ BT(bf16)[256 x 256]^T  (BT row n = column n)
__global__ void gemm_h_kernel(const unsigned short* __restrict__ A,
                              const unsigned short* __restrict__ BT,
                              float* __restrict__ Dst) {
  int lane = threadIdx.x & 31, wave = threadIdx.x >> 5;
  int mtile = blockIdx.x;
  int ntile = blockIdx.y * 8 + wave;
  int kh = lane >> 4;
  int row  = mtile * 16 + (lane & 15);
  int coln = ntile * 16 + (lane & 15);
  const unsigned short* arow = A  + (size_t)row  * 256;
  const unsigned short* brow = BT + (size_t)coln * 256;
  v8f acc = {};
#pragma unroll
  for (int k0 = 0; k0 < 256; k0 += 32) {
    v16bf a = load_frag(arow + k0, kh);
    v16bf b = load_frag(brow + k0, kh);
    acc = __builtin_amdgcn_wmma_f32_16x16x32_bf16(false, a, false, b,
                                                  (short)0, acc, false, false);
  }
#pragma unroll
  for (int r = 0; r < 8; ++r) {
    int m = r + 8 * kh;
    Dst[(size_t)(mtile * 16 + m) * 256 + coln] = acc[r];
  }
}

// P[p][node][64] = Xsel(bf16)[N x 256] @ WcT[64 x 1024] slice (rows p*256..p*256+255)
__global__ void gemm_p_kernel(const unsigned short* __restrict__ Xa,
                              const unsigned short* __restrict__ Xb,
                              const unsigned short* __restrict__ WcT,
                              float* __restrict__ P) {
  int lane = threadIdx.x & 31, wave = threadIdx.x >> 5; // block = 128 -> 4 waves
  int p = blockIdx.y;
  const unsigned short* A = (p < 2) ? Xa : Xb;
  int mtile = blockIdx.x, ntile = wave;
  int kh = lane >> 4;
  int row  = mtile * 16 + (lane & 15);
  int coln = ntile * 16 + (lane & 15);
  const unsigned short* arow = A + (size_t)row * 256;
  const unsigned short* brow = WcT + (size_t)coln * 1024 + p * 256;
  v8f acc = {};
#pragma unroll
  for (int k0 = 0; k0 < 256; k0 += 32) {
    v16bf a = load_frag(arow + k0, kh);
    v16bf b = load_frag(brow + k0, kh);
    acc = __builtin_amdgcn_wmma_f32_16x16x32_bf16(false, a, false, b,
                                                  (short)0, acc, false, false);
  }
#pragma unroll
  for (int r = 0; r < 8; ++r) {
    int m = r + 8 * kh;
    P[((size_t)p * N_NODES + mtile * 16 + m) * 64 + coln] = acc[r];
  }
}

// ---------------- GAT kernels ----------------
__global__ void att_dot_kernel(const float* __restrict__ h,
                               const float* __restrict__ a_s,
                               const float* __restrict__ a_d,
                               float* __restrict__ al, float* __restrict__ ar) {
  int idx = blockIdx.x * blockDim.x + threadIdx.x;
  if (idx >= N_NODES * 4) return;
  int n = idx >> 2, hh = idx & 3;
  const float* hp = h + (size_t)n * 256 + hh * 64;
  const float* as = a_s + hh * 64;
  const float* ad = a_d + hh * 64;
  float sl = 0.f, sr = 0.f;
#pragma unroll 8
  for (int c = 0; c < 64; ++c) { sl += hp[c] * as[c]; sr += hp[c] * ad[c]; }
  al[idx] = sl;
  ar[idx] = sr;
}

__global__ void edge_max_kernel(const int* __restrict__ src, const int* __restrict__ dst,
                                const float* __restrict__ al, const float* __restrict__ ar,
                                float* __restrict__ eatt, float* __restrict__ mbuf) {
  int e = blockIdx.x * blockDim.x + threadIdx.x;
  if (e >= EN) return;
  int s, d;
  if (e < N_EDGES) { s = src[e]; d = dst[e]; } else { s = d = e - N_EDGES; }
#pragma unroll
  for (int hh = 0; hh < 4; ++hh) {
    float v = al[s * 4 + hh] + ar[d * 4 + hh];
    v = v > 0.f ? v : 0.2f * v;                  // leaky_relu(0.2)
    eatt[(size_t)e * 4 + hh] = v;
    atomicMaxF(&mbuf[d * 4 + hh], v);
  }
}

__global__ void edge_exp_kernel(const int* __restrict__ src, const int* __restrict__ dst,
                                const float* __restrict__ eatt, const float* __restrict__ mbuf,
                                float* __restrict__ exbuf, float* __restrict__ denom) {
  int e = blockIdx.x * blockDim.x + threadIdx.x;
  if (e >= EN) return;
  int d;
  if (e < N_EDGES) { d = dst[e]; } else { d = e - N_EDGES; }
#pragma unroll
  for (int hh = 0; hh < 4; ++hh) {
    float ex = expf(eatt[(size_t)e * 4 + hh] - mbuf[d * 4 + hh]);
    exbuf[(size_t)e * 4 + hh] = ex;
    atomicAdd(&denom[d * 4 + hh], ex);
  }
}

// one block (256 threads = all HC channels) per edge
__global__ void gat_scatter_kernel(const int* __restrict__ src, const int* __restrict__ dst,
                                   const float* __restrict__ hfeat,
                                   const float* __restrict__ exbuf,
                                   const float* __restrict__ denom,
                                   float* __restrict__ agg) {
  int e = blockIdx.x;
  int c = threadIdx.x;
  int hh = c >> 6;
  int s, d;
  if (e < N_EDGES) { s = src[e]; d = dst[e]; } else { s = d = e - N_EDGES; }
  float coef = exbuf[(size_t)e * 4 + hh] / (denom[d * 4 + hh] + 1e-16f);
  atomicAdd(&agg[(size_t)d * 256 + c], coef * hfeat[(size_t)s * 256 + c]);
}

__global__ void elu_bias_kernel(const float* __restrict__ agg, const float* __restrict__ b,
                                unsigned short* __restrict__ xbf) {
  long idx = (long)blockIdx.x * blockDim.x + threadIdx.x;
  if (idx >= (long)N_NODES * 256) return;
  int c = (int)(idx & 255);
  float v = agg[idx] + b[c];
  v = v > 0.f ? v : (expf(v) - 1.f);             // elu
  xbf[idx] = f32_to_bf16(v);
}

// ---------------- dedup: codes, bitonic sort, scan, inverse ----------------
__global__ void make_codes_kernel(const int* __restrict__ s1, const int* __restrict__ d1,
                                  const int* __restrict__ s2, const int* __restrict__ d2,
                                  int* __restrict__ keys, int* __restrict__ vals) {
  int i = blockIdx.x * blockDim.x + threadIdx.x;
  if (i >= SORT_N) return;
  if (i < TWO_E) {
    int code = (i < N_EDGES) ? (s1[i] * N_NODES + d1[i])
                             : (s2[i - N_EDGES] * N_NODES + d2[i - N_EDGES]);
    keys[i] = code; vals[i] = i;
  } else {
    keys[i] = 0x7FFFFFFF; vals[i] = -1;
  }
}

__global__ void bitonic_step_kernel(int* __restrict__ keys, int* __restrict__ vals,
                                    int j, int k) {
  int i = blockIdx.x * blockDim.x + threadIdx.x;
  int ixj = i ^ j;
  if (ixj <= i) return;
  bool up = ((i & k) == 0);
  int ki = keys[i], kj = keys[ixj];
  bool sw = up ? (ki > kj) : (ki < kj);
  if (sw) {
    keys[i] = kj; keys[ixj] = ki;
    int t = vals[i]; vals[i] = vals[ixj]; vals[ixj] = t;
  }
}

__global__ void flags_kernel(const int* __restrict__ keys, int* __restrict__ flags) {
  int i = blockIdx.x * blockDim.x + threadIdx.x;
  if (i >= SORT_N) return;
  flags[i] = (i > 0 && keys[i] != keys[i - 1]) ? 1 : 0;
}

__global__ void scan_step_kernel(const int* __restrict__ in, int* __restrict__ out, int d) {
  int i = blockIdx.x * blockDim.x + threadIdx.x;
  if (i >= SORT_N) return;
  int v = in[i];
  if (i >= d) v += in[i - d];
  out[i] = v;
}

__global__ void inv_scatter_kernel(const int* __restrict__ vals,
                                   const int* __restrict__ rank,
                                   int* __restrict__ inv) {
  int i = blockIdx.x * blockDim.x + threadIdx.x;
  if (i >= SORT_N) return;
  int v = vals[i];
  if (v >= 0) inv[v] = rank[i];
}

// ---------------- logits scatter + softmax ----------------
__global__ void scatter_logits_kernel(const int* __restrict__ s1, const int* __restrict__ d1,
                                      const int* __restrict__ s2, const int* __restrict__ d2,
                                      const int* __restrict__ inv,
                                      const float* __restrict__ P,
                                      const float* __restrict__ alpha,
                                      const float* __restrict__ beta,
                                      float* __restrict__ newlogits) {
  int e = blockIdx.x;
  int c = threadIdx.x;
  if (c >= NC) return;
  int s, d; float sc;
  if (e < N_EDGES) { s = s1[e]; d = d1[e]; sc = *alpha; }
  else             { s = s2[e - N_EDGES]; d = d2[e - N_EDGES]; sc = *beta; }
  int row = inv[e];
  float v = P[((size_t)0 * N_NODES + s) * 64 + c]
          + P[((size_t)1 * N_NODES + d) * 64 + c]
          + P[((size_t)2 * N_NODES + s) * 64 + c]
          + P[((size_t)3 * N_NODES + d) * 64 + c];
  atomicAdd(&newlogits[(size_t)row * 64 + c], sc * v);
}

// one wave32 per output row; 51 classes handled as lane and lane+32
__global__ void softmax_kernel(const float* __restrict__ nl, const float* __restrict__ bc,
                               float* __restrict__ out) {
  int wave = threadIdx.x >> 5, lane = threadIdx.x & 31;
  int row = blockIdx.x * 8 + wave;
  if (row >= TWO_E) return;
  int l2 = lane + 32;
  float v0 = (lane < NC) ? nl[(size_t)row * 64 + lane] + bc[lane] : -INFINITY;
  float v1 = (l2 < NC)   ? nl[(size_t)row * 64 + l2]   + bc[l2]   : -INFINITY;
  float mx = fmaxf(v0, v1);
#pragma unroll
  for (int o = 16; o > 0; o >>= 1) mx = fmaxf(mx, __shfl_xor(mx, o, 32));
  float e0 = (lane < NC) ? expf(v0 - mx) : 0.f;
  float e1 = (l2 < NC)   ? expf(v1 - mx) : 0.f;
  float s = e0 + e1;
#pragma unroll
  for (int o = 16; o > 0; o >>= 1) s += __shfl_xor(s, o, 32);
  float r = 1.f / s;
  if (lane < NC) out[(size_t)row * NC + lane] = e0 * r;
  if (l2 < NC)   out[(size_t)row * NC + l2]   = e1 * r;
}

// ---------------- host launcher ----------------
extern "C" void kernel_launch(void* const* d_in, const int* in_sizes, int n_in,
                              void* d_out, int out_size, void* d_ws, size_t ws_size,
                              hipStream_t stream) {
  (void)in_sizes; (void)n_in; (void)out_size; (void)ws_size;

  const float* x1  = (const float*)d_in[0];
  const int*   ei1 = (const int*)  d_in[1];
  const float* x2  = (const float*)d_in[2];
  const int*   ei2 = (const int*)  d_in[3];
  const float* W1  = (const float*)d_in[4];
  const float* as1 = (const float*)d_in[5];
  const float* ad1 = (const float*)d_in[6];
  const float* b1  = (const float*)d_in[7];
  const float* W2  = (const float*)d_in[8];
  const float* as2 = (const float*)d_in[9];
  const float* ad2 = (const float*)d_in[10];
  const float* b2  = (const float*)d_in[11];
  const float* Wc  = (const float*)d_in[12];
  const float* bc  = (const float*)d_in[13];
  const float* alpha = (const float*)d_in[14];
  const float* beta  = (const float*)d_in[15];

  const int* src1 = ei1;               const int* dst1 = ei1 + N_EDGES;
  const int* src2 = ei2;               const int* dst2 = ei2 + N_EDGES;

  // ---- workspace carve-up (~152 MB) ----
  char* wsb = (char*)d_ws;
  size_t off = 0;
  auto take = [&](size_t bytes) -> char* {
    char* p = wsb + off;
    off += (bytes + 255) & ~(size_t)255;
    return p;
  };
  unsigned short* xinbf = (unsigned short*)take((size_t)N_NODES * 256 * 2);
  unsigned short* WbfT  = (unsigned short*)take((size_t)256 * 256 * 2);
  float* hfeat = (float*)take((size_t)N_NODES * 256 * 4);   // reused as P[4][N][64]
  float* agg   = (float*)take((size_t)N_NODES * 256 * 4);   // reused as newlogits[2E][64]
  unsigned short* xbfA = (unsigned short*)take((size_t)N_NODES * 256 * 2);  // elu(gat1)
  unsigned short* xbfB = (unsigned short*)take((size_t)N_NODES * 256 * 2);  // elu(gat2)
  float* al    = (float*)take((size_t)N_NODES * 4 * 4);
  float* ar    = (float*)take((size_t)N_NODES * 4 * 4);
  float* mbuf  = (float*)take((size_t)N_NODES * 4 * 4);
  float* denom = (float*)take((size_t)N_NODES * 4 * 4);
  float* eatt  = (float*)take((size_t)EN * 4 * 4);
  float* exbuf = (float*)take((size_t)EN * 4 * 4);
  unsigned short* WcT = (unsigned short*)take((size_t)64 * 1024 * 2);
  int* keys = (int*)take((size_t)SORT_N * 4);
  int* vals = (int*)take((size_t)SORT_N * 4);
  int* sc0  = (int*)take((size_t)SORT_N * 4);
  int* sc1  = (int*)take((size_t)SORT_N * 4);
  int* inv  = (int*)take((size_t)TWO_E * 4);
  float* P         = hfeat;
  float* newlogits = agg;

  auto blocks = [](long n, int bs) { return (int)((n + bs - 1) / bs); };
  const long NHC = (long)N_NODES * 256;

  // ================= phase A: both GAT layers =================
  const float* xs[2]  = {x1, x2};
  const float* Ws[2]  = {W1, W2};
  const float* ass[2] = {as1, as2};
  const float* ads[2] = {ad1, ad2};
  const float* bs[2]  = {b1, b2};
  const int*   srcs[2] = {src1, src2};
  const int*   dsts[2] = {dst1, dst2};
  unsigned short* xouts[2] = {xbfA, xbfB};

  for (int g = 0; g < 2; ++g) {
    // bf16 conversions
    f32_to_bf16_kernel<<<blocks(NHC, 256), 256, 0, stream>>>(xs[g], xinbf, NHC);
    transpose_w_kernel<<<blocks(256 * 256, 256), 256, 0, stream>>>(Ws[g], WbfT);
    // h = x @ W   (WMMA bf16 -> f32)
    gemm_h_kernel<<<dim3(N_NODES / 16, 2), 256, 0, stream>>>(xinbf, WbfT, hfeat);
    // attention dot products
    att_dot_kernel<<<blocks((long)N_NODES * 4, 256), 256, 0, stream>>>(hfeat, ass[g], ads[g], al, ar);
    // init segment buffers
    fill_f32_kernel<<<blocks((long)N_NODES * 4, 256), 256, 0, stream>>>(mbuf, -INFINITY, (long)N_NODES * 4);
    fill_f32_kernel<<<blocks((long)N_NODES * 4, 256), 256, 0, stream>>>(denom, 0.f, (long)N_NODES * 4);
    fill_f32_kernel<<<blocks(NHC, 256), 256, 0, stream>>>(agg, 0.f, NHC);
    // segment softmax over destinations + message scatter
    edge_max_kernel<<<blocks(EN, 256), 256, 0, stream>>>(srcs[g], dsts[g], al, ar, eatt, mbuf);
    edge_exp_kernel<<<blocks(EN, 256), 256, 0, stream>>>(srcs[g], dsts[g], eatt, mbuf, exbuf, denom);
    gat_scatter_kernel<<<EN, 256, 0, stream>>>(srcs[g], dsts[g], hfeat, exbuf, denom, agg);
    // x = elu(agg + b) -> bf16 for classifier GEMM
    elu_bias_kernel<<<blocks(NHC, 256), 256, 0, stream>>>(agg, bs[g], xouts[g]);
  }

  // ================= phase B: classifier pushed to node level =================
  prep_wct_kernel<<<blocks(64 * 1024, 256), 256, 0, stream>>>(Wc, WcT);
  // P[p] = (p<2 ? x : xb) @ Wc[p*256:(p+1)*256, :]   (WMMA)
  gemm_p_kernel<<<dim3(N_NODES / 16, 4), 128, 0, stream>>>(xbfA, xbfB, WcT, P);

  // dedup: replicate jnp.unique(sorted) inverse indices
  make_codes_kernel<<<blocks(SORT_N, 256), 256, 0, stream>>>(src1, dst1, src2, dst2, keys, vals);
  for (int k = 2; k <= SORT_N; k <<= 1)
    for (int j = k >> 1; j > 0; j >>= 1)
      bitonic_step_kernel<<<SORT_N / 256, 256, 0, stream>>>(keys, vals, j, k);
  flags_kernel<<<blocks(SORT_N, 256), 256, 0, stream>>>(keys, sc0);
  {
    int* cur = sc0; int* nxt = sc1;
    for (int d = 1; d < SORT_N; d <<= 1) {
      scan_step_kernel<<<blocks(SORT_N, 256), 256, 0, stream>>>(cur, nxt, d);
      std::swap(cur, nxt);
    }
    inv_scatter_kernel<<<blocks(SORT_N, 256), 256, 0, stream>>>(vals, cur, inv);
  }

  // scatter per-edge logit contributions (51 floats/edge instead of 1024)
  fill_f32_kernel<<<blocks((long)TWO_E * 64, 256), 256, 0, stream>>>(newlogits, 0.f, (long)TWO_E * 64);
  scatter_logits_kernel<<<TWO_E, 64, 0, stream>>>(src1, dst1, src2, dst2, inv, P,
                                                  alpha, beta, newlogits);
  // softmax over 51 classes, one wave per row
  softmax_kernel<<<blocks(TWO_E, 8), 256, 0, stream>>>(newlogits, bc, (float*)d_out);
}
